// AxialMHA_18940805775987
// MI455X (gfx1250) — matrix operand order repaired and checked by reading, not compiled, not run on hardware
//
#include <hip/hip_runtime.h>
#include <hip/hip_bf16.h>
#include <math.h>

#define D_MODEL 1024
#define NHEAD   16
#define D_HEAD  64
#define SEQ     2048
#define BATCH   2

typedef __attribute__((ext_vector_type(16))) _Float16 v16h;
typedef __attribute__((ext_vector_type(8)))  _Float16 v8h;
typedef __attribute__((ext_vector_type(8)))  float    v8f;

static __device__ __forceinline__ v8f wmma_f16(v16h a, v16h b, v8f c) {
  // D = A(16x32 f16) x B(32x16 f16) + C(16x16 f32)
  return __builtin_amdgcn_wmma_f32_16x16x32_f16(false, a, false, b, (short)0, c,
                                                false, false);
}

// ---- Async global->LDS staging (CDNA5 ASYNCcnt path) -----------------------
static __device__ __forceinline__ void async_copy16(unsigned lds_addr,
                                                    unsigned long long gaddr) {
  asm volatile("global_load_async_to_lds_b128 %0, %1, off"
               :: "v"(lds_addr), "v"(gaddr) : "memory");
}
static __device__ __forceinline__ void wait_async_le8() {
  asm volatile("s_wait_asynccnt 0x8" ::: "memory");
}
static __device__ __forceinline__ void wait_async_0() {
  asm volatile("s_wait_asynccnt 0x0" ::: "memory");
}

// ---- Fragment loaders (layouts per CDNA5 ISA 7.12.2) -----------------------
// A (16x32, 16-bit): lanes 0-15 -> rows, halves K={0..7,16..23};
//                    lanes 16-31 -> rows, halves K={8..15,24..31}.
static __device__ __forceinline__ v16h load_a_f32(const float* base, int ld,
                                                  int m0, int k0, int lane) {
  const int row  = m0 + (lane & 15);
  const int ksel = (lane >> 4) & 1;
  const float* p0 = base + (size_t)row * ld + k0 + ksel * 8;
  const float* p1 = p0 + 16;
  v16h a;
#pragma unroll
  for (int j = 0; j < 8; ++j) { a[j] = (_Float16)p0[j]; a[j + 8] = (_Float16)p1[j]; }
  return a;
}

static __device__ __forceinline__ v16h load_a_f16(const _Float16* base, int ld,
                                                  int m0, int k0, int lane) {
  const int row  = m0 + (lane & 15);
  const int ksel = (lane >> 4) & 1;
  const _Float16* p0 = base + (size_t)row * ld + k0 + ksel * 8;
  const _Float16* p1 = p0 + 16;
  v16h a;
#pragma unroll
  for (int j = 0; j < 8; ++j) { a[j] = p0[j]; a[j + 8] = p1[j]; }
  return a;
}

// B (32x16, 16-bit): lanes -> N column (lane%16), halves K={0..7,16..23} / {8..15,24..31}.
static __device__ __forceinline__ v16h load_b_f32(const float* base, int ld,
                                                  int k0, int n0, int lane) {
  const float* p = base + (size_t)(k0 + ((lane >> 4) << 3)) * ld + n0 + (lane & 15);
  v16h b;
#pragma unroll
  for (int j = 0; j < 8; ++j) { b[j] = (_Float16)p[(size_t)j * ld]; b[j + 8] = (_Float16)p[(size_t)(j + 16) * ld]; }
  return b;
}

static __device__ __forceinline__ v16h load_b_f16(const _Float16* base, int ld,
                                                  int k0, int n0, int lane) {
  const _Float16* p = base + (size_t)(k0 + ((lane >> 4) << 3)) * ld + n0 + (lane & 15);
  v16h b;
#pragma unroll
  for (int j = 0; j < 8; ++j) { b[j] = p[(size_t)j * ld]; b[j + 8] = p[(size_t)(j + 16) * ld]; }
  return b;
}

// B fragment for S = Q*K^T from d-major K^T[64][SEQ]: lane%16 = key,
// halves = d indices {0..7,16..23} / {8..15,24..31}.
static __device__ __forceinline__ v16h load_bKt(const _Float16* KtH, int c0,
                                                int n0, int lane) {
  const _Float16* p = KtH + (size_t)(c0 + ((lane >> 4) << 3)) * SEQ + n0 + (lane & 15);
  v16h b;
#pragma unroll
  for (int j = 0; j < 8; ++j) { b[j] = p[(size_t)j * SEQ]; b[j + 8] = p[(size_t)(j + 16) * SEQ]; }
  return b;
}

// ---- Kernel 1: QKV = x @ Wqkv + bqkv; Q,V head-major, K transposed ---------
__global__ __launch_bounds__(256) void qkv_kernel(const float* __restrict__ x,
                                                  const float* __restrict__ W,
                                                  const float* __restrict__ bias,
                                                  _Float16* __restrict__ Qo,
                                                  _Float16* __restrict__ Kt,
                                                  _Float16* __restrict__ Vo) {
  const int lane = threadIdx.x & 31;
  const int wave = threadIdx.x >> 5;
  const int w    = blockIdx.x * 8 + wave;     // 256 M-tiles * 48 N-tiles
  const int mt   = w / 48;
  const int nt   = w % 48;
  const int m0   = mt * 16;
  const int col0 = nt * 64;                   // 64-aligned -> one head

  v8f c[4];
#pragma unroll
  for (int g = 0; g < 4; ++g) {
    const float bv = bias[col0 + g * 16 + (lane & 15)];
#pragma unroll
    for (int r = 0; r < 8; ++r) c[g][r] = bv;
  }

  for (int k0 = 0; k0 < D_MODEL; k0 += 32) {
    if (k0 + 32 < D_MODEL)                     // prefetch next W k-slab
      __builtin_prefetch(W + (size_t)(k0 + 32 + lane) * (3 * D_MODEL) + col0, 0, 0);
    const v16h a = load_a_f32(x, D_MODEL, m0, k0, lane);
#pragma unroll
    for (int g = 0; g < 4; ++g) {
      const v16h bf = load_b_f32(W, 3 * D_MODEL, k0, col0 + g * 16, lane);
      c[g] = wmma_f16(a, bf, c[g]);
    }
  }

  const int part = col0 / D_MODEL;            // 0=Q 1=K 2=V
  const int rem  = col0 % D_MODEL;
  const int h    = rem / D_HEAD;
  const int bb   = m0 >> 11;                  // batch (tile never crosses)
  if (part == 1) {
    // K stored d-major: Kt[b][h][dh][n]; rows contiguous across accum regs
    const int n0 = (m0 & (SEQ - 1)) + ((lane >> 4) << 3);
#pragma unroll
    for (int g = 0; g < 4; ++g) {
      const int dh = g * 16 + (lane & 15);
      v8h pk;
#pragma unroll
      for (int r = 0; r < 8; ++r) pk[r] = (_Float16)c[g][r];
      *(v8h*)&Kt[(((size_t)bb * NHEAD + h) * D_HEAD + dh) * SEQ + n0] = pk;
    }
  } else {
    _Float16* dst = (part == 0) ? Qo : Vo;
#pragma unroll
    for (int g = 0; g < 4; ++g) {
#pragma unroll
      for (int r = 0; r < 8; ++r) {
        const int m  = m0 + r + ((lane >> 4) << 3);
        const int n  = m & (SEQ - 1);
        const int dh = g * 16 + (lane & 15);
        dst[(((size_t)bb * NHEAD + h) * SEQ + n) * D_HEAD + dh] = (_Float16)c[g][r];
      }
    }
  }
}

// ---- Kernel 2: flash attention; async double-buffered V tiles in LDS -------
__global__ __launch_bounds__(256) void attn_kernel(const _Float16* __restrict__ Q,
                                                   const _Float16* __restrict__ Kt,
                                                   const _Float16* __restrict__ V,
                                                   _Float16* __restrict__ Y) {
  __shared__ _Float16 Plds[8][16][32];            // P staging, 8 KB
  __shared__ _Float16 Vlds[8][2][32 * D_HEAD];    // V double-buffer, 64 KB
  const int lane = threadIdx.x & 31;
  const int wave = threadIdx.x >> 5;
  const int w    = blockIdx.x * 8 + wave;         // 4096 waves total
  const int qt   = w & 127;
  const int h    = (w >> 7) & (NHEAD - 1);
  const int b    = w >> 11;

  const size_t headOff = ((size_t)b * NHEAD + h) * SEQ * D_HEAD;
  const _Float16* Qb  = Q + headOff;
  const _Float16* KtH = Kt + headOff;             // [64][SEQ] d-major
  const _Float16* Vb  = V + headOff;
  const int m0 = qt * 16;

  const v16h qa0 = load_a_f16(Qb, D_HEAD, m0, 0, lane);
  const v16h qa1 = load_a_f16(Qb, D_HEAD, m0, 32, lane);

  v8f o0 = 0.f, o1 = 0.f, o2 = 0.f, o3 = 0.f;
  float mi[8], li[8];
#pragma unroll
  for (int r = 0; r < 8; ++r) { mi[r] = -INFINITY; li[r] = 0.f; }
  const float scale = 0.125f;                     // 1/sqrt(64)

  // prime buffer 0 with V tile 0 (32 rows x 64 cols = 4KB = 8 x 512B)
  {
    const unsigned long long g0 = (unsigned long long)(uintptr_t)Vb;
    const unsigned l0 = (unsigned)(uintptr_t)&Vlds[wave][0][0];
#pragma unroll
    for (int i = 0; i < 8; ++i)
      async_copy16(l0 + i * 512 + lane * 16, g0 + i * 512 + lane * 16);
  }

  for (int it = 0; it < SEQ / 32; ++it) {
    const int kt = it * 32;
    const int buf = it & 1;
    const bool more = (it + 1) < (SEQ / 32);
    if (more) {                                   // async-stage next V tile
      const unsigned long long g0 =
          (unsigned long long)(uintptr_t)(Vb + (size_t)(kt + 32) * D_HEAD);
      const unsigned l0 = (unsigned)(uintptr_t)&Vlds[wave][buf ^ 1][0];
#pragma unroll
      for (int i = 0; i < 8; ++i)
        async_copy16(l0 + i * 512 + lane * 16, g0 + i * 512 + lane * 16);
      __builtin_prefetch(KtH + (size_t)lane * SEQ + kt + 32, 0, 0);
    }

    // S = Q K^T over 32 keys (B-frag: per d row, 16 lanes read 32 contiguous bytes)
    v8f s0 = 0.f, s1 = 0.f;
    s0 = wmma_f16(qa0, load_bKt(KtH,  0, kt,      lane), s0);
    s0 = wmma_f16(qa1, load_bKt(KtH, 32, kt,      lane), s0);
    s1 = wmma_f16(qa0, load_bKt(KtH,  0, kt + 16, lane), s1);
    s1 = wmma_f16(qa1, load_bKt(KtH, 32, kt + 16, lane), s1);

    float alpha[8];
#pragma unroll
    for (int r = 0; r < 8; ++r) {
      const float v0 = s0[r] * scale;
      const float v1 = s1[r] * scale;
      float mx = fmaxf(v0, v1);
      mx = fmaxf(mx, __shfl_xor(mx, 1, 16));
      mx = fmaxf(mx, __shfl_xor(mx, 2, 16));
      mx = fmaxf(mx, __shfl_xor(mx, 4, 16));
      mx = fmaxf(mx, __shfl_xor(mx, 8, 16));
      const float mnew = fmaxf(mi[r], mx);
      const float a    = __expf(mi[r] - mnew);
      const float p0   = __expf(v0 - mnew);
      const float p1   = __expf(v1 - mnew);
      float ps = p0 + p1;
      ps += __shfl_xor(ps, 1, 16);
      ps += __shfl_xor(ps, 2, 16);
      ps += __shfl_xor(ps, 4, 16);
      ps += __shfl_xor(ps, 8, 16);
      li[r] = li[r] * a + ps;
      mi[r] = mnew;
      alpha[r] = a;
      const int m = r + ((lane >> 4) << 3);
      Plds[wave][m][(lane & 15)]      = (_Float16)p0;
      Plds[wave][m][(lane & 15) + 16] = (_Float16)p1;
    }
#pragma unroll
    for (int r = 0; r < 8; ++r) {
      o0[r] *= alpha[r]; o1[r] *= alpha[r];
      o2[r] *= alpha[r]; o3[r] *= alpha[r];
    }

    // current V tile is complete once ASYNCcnt <= (in-flight next tile = 8)
    if (more) wait_async_le8(); else wait_async_0();

    const v16h pa = load_a_f16(&Plds[wave][0][0], 32, 0, 0, lane);
    const _Float16* vt = &Vlds[wave][buf][0];
    o0 = wmma_f16(pa, load_b_f16(vt, D_HEAD, 0,  0, lane), o0);
    o1 = wmma_f16(pa, load_b_f16(vt, D_HEAD, 0, 16, lane), o1);
    o2 = wmma_f16(pa, load_b_f16(vt, D_HEAD, 0, 32, lane), o2);
    o3 = wmma_f16(pa, load_b_f16(vt, D_HEAD, 0, 48, lane), o3);
  }

  // epilogue: normalize and write Y[b*SEQ+row][h*64 + col] as f16
#pragma unroll
  for (int r = 0; r < 8; ++r) {
    const float inv = 1.0f / li[r];
    const int row   = m0 + r + ((lane >> 4) << 3);
    const size_t bn = (size_t)b * SEQ + row;
    _Float16* yrow  = Y + bn * D_MODEL + h * D_HEAD;
    yrow[ 0 + (lane & 15)] = (_Float16)(o0[r] * inv);
    yrow[16 + (lane & 15)] = (_Float16)(o1[r] * inv);
    yrow[32 + (lane & 15)] = (_Float16)(o2[r] * inv);
    yrow[48 + (lane & 15)] = (_Float16)(o3[r] * inv);
  }
}

// ---- Kernel 3: out = Y @ Wproj + bproj (f32 out) ---------------------------
__global__ __launch_bounds__(256) void proj_kernel(const _Float16* __restrict__ Y,
                                                   const float* __restrict__ W,
                                                   const float* __restrict__ bias,
                                                   float* __restrict__ out) {
  const int lane = threadIdx.x & 31;
  const int wave = threadIdx.x >> 5;
  const int w    = blockIdx.x * 8 + wave;     // 256 M-tiles * 16 N-tiles
  const int mt   = w / 16;
  const int nt   = w % 16;
  const int m0   = mt * 16;
  const int col0 = nt * 64;

  v8f c[4];
#pragma unroll
  for (int g = 0; g < 4; ++g) {
    const float bv = bias[col0 + g * 16 + (lane & 15)];
#pragma unroll
    for (int r = 0; r < 8; ++r) c[g][r] = bv;
  }

  for (int k0 = 0; k0 < D_MODEL; k0 += 32) {
    if (k0 + 32 < D_MODEL)
      __builtin_prefetch(W + (size_t)(k0 + 32 + lane) * D_MODEL + col0, 0, 0);
    const v16h a = load_a_f16(Y, D_MODEL, m0, k0, lane);
#pragma unroll
    for (int g = 0; g < 4; ++g) {
      const v16h bf = load_b_f32(W, D_MODEL, k0, col0 + g * 16, lane);
      c[g] = wmma_f16(a, bf, c[g]);
    }
  }

#pragma unroll
  for (int g = 0; g < 4; ++g) {
#pragma unroll
    for (int r = 0; r < 8; ++r) {
      const int m = m0 + r + ((lane >> 4) << 3);
      out[(size_t)m * D_MODEL + col0 + g * 16 + (lane & 15)] = c[g][r];
    }
  }
}

extern "C" void kernel_launch(void* const* d_in, const int* in_sizes, int n_in,
                              void* d_out, int out_size, void* d_ws, size_t ws_size,
                              hipStream_t stream) {
  (void)in_sizes; (void)n_in; (void)out_size; (void)ws_size;
  const float* x     = (const float*)d_in[0];
  const float* Wqkv  = (const float*)d_in[1];
  const float* bqkv  = (const float*)d_in[2];
  const float* Wproj = (const float*)d_in[3];
  const float* bproj = (const float*)d_in[4];
  float* out = (float*)d_out;

  const size_t headElems = (size_t)BATCH * NHEAD * SEQ * D_HEAD;  // 4M halves
  _Float16* Q  = (_Float16*)d_ws;
  _Float16* Kt = Q + headElems;   // transposed per head: [B,H,64,SEQ]
  _Float16* V  = Kt + headElems;
  _Float16* Y  = V + headElems;   // [B*SEQ, D_MODEL] f16

  // 256 M-tiles * 48 N-tiles = 12288 waves / 8 per block
  qkv_kernel<<<1536, 256, 0, stream>>>(x, Wqkv, bqkv, Q, Kt, V);
  // B*H*(SEQ/16) = 4096 waves / 8 per block
  attn_kernel<<<512, 256, 0, stream>>>(Q, Kt, V, Y);
  // 256 M-tiles * 16 N-tiles = 4096 waves / 8 per block
  proj_kernel<<<512, 256, 0, stream>>>(Y, Wproj, bproj, out);
}
